// StyleLossW2_51573967290833
// MI455X (gfx1250) — matrix operand order repaired and checked by reading, not compiled, not run on hardware
//
#include <hip/hip_runtime.h>

#define BB   8
#define CC   512
#define HW_  16384
#define EPSV 1e-4f

typedef __attribute__((ext_vector_type(16))) __bf16 v16bf;
typedef __attribute__((ext_vector_type(8)))  float  v8f;

struct Frag { v16bf hi; v16bf lo; };

// Split 16 f32 values into bf16 hi/lo fragments (bf16x3 trick), and accumulate
// their f32 sum (used for the fused channel-mean in the Gram pass).
__device__ __forceinline__ Frag make_frag(float4 v0, float4 v1, float4 v2, float4 v3,
                                          float& sum) {
  float f[16] = {v0.x, v0.y, v0.z, v0.w, v1.x, v1.y, v1.z, v1.w,
                 v2.x, v2.y, v2.z, v2.w, v3.x, v3.y, v3.z, v3.w};
  Frag r;
  float s = 0.f;
#pragma unroll
  for (int e = 0; e < 16; ++e) {
    float x = f[e];
    s += x;
    __bf16 h = (__bf16)x;
    r.hi[e] = h;
    r.lo[e] = (__bf16)(x - (float)h);
  }
  sum += s;
  return r;
}

__device__ __forceinline__ v8f wmma3(const Frag& a, const Frag& b, v8f c) {
  c = __builtin_amdgcn_wmma_f32_16x16x32_bf16(false, a.hi, false, b.hi, (short)0, c, false, false);
  c = __builtin_amdgcn_wmma_f32_16x16x32_bf16(false, a.hi, false, b.lo, (short)0, c, false, false);
  c = __builtin_amdgcn_wmma_f32_16x16x32_bf16(false, a.lo, false, b.hi, (short)0, c, false, false);
  return c;
}

// C[m][n] = epilogue( sum_k A[m][k] * Bt[n][k] )
// Bt is the transpose of the logical B operand; every B in this problem is
// symmetric (or is X itself for the Gram), so Bt rows are contiguous.
// mode 0: gram  -> C = acc*scale - mean[m]*mean[n] + EPS*(m==n); writes means to aux
// mode 1: NS t  -> C = 0.5*(3*(m==n) - acc)
// mode 2: plain -> C = acc
// mode 3: C = acc * aux[b]   (per-batch scale, folds tcs = sqrt(n)*y into sandwich)
__device__ __forceinline__ void gemm_body(const float* __restrict__ Ab,
                                          const float* __restrict__ Bb,
                                          float* __restrict__ Cb,
                                          float* __restrict__ aux,
                                          int b, int K, int lda,
                                          float scale, int mode)
{
  const int lane = threadIdx.x & 31;
  const int wid  = threadIdx.x >> 5;
  const int bm   = blockIdx.x * 64 + (wid >> 1) * 32;
  const int bn   = blockIdx.y * 64 + (wid & 1) * 32;
  const int mrow = lane & 15;
  const int hi16 = lane >> 4;

  v8f acc00 = {}, acc01 = {}, acc10 = {}, acc11 = {};
  float sa0 = 0.f, sa1 = 0.f, sb0 = 0.f, sb1 = 0.f;

  // A-fragment lane layout (ISA 7.12.2, 16-bit A 16x32):
  //   lanes 0-15: row m, K = [0..7],[16..23]; lanes 16-31: K = [8..15],[24..31]
  // B-fragment (32x16): same K order as A: lanes 0-15 K=[0..7],[16..23]; lanes 16-31 K=[8..15],[24..31].
  const int koffA = hi16 * 8;
  const int koffB = hi16 * 8;
  const float* pa0 = Ab + (long)(bm + mrow)      * lda + koffA;
  const float* pa1 = Ab + (long)(bm + 16 + mrow) * lda + koffA;
  const float* pb0 = Bb + (long)(bn + mrow)      * lda + koffB;
  const float* pb1 = Bb + (long)(bn + 16 + mrow) * lda + koffB;

#pragma unroll 2
  for (int k0 = 0; k0 < K; k0 += 32) {
    const float4* qa0 = (const float4*)(pa0 + k0);
    const float4* qa1 = (const float4*)(pa1 + k0);
    const float4* qb0 = (const float4*)(pb0 + k0);
    const float4* qb1 = (const float4*)(pb1 + k0);
    Frag a0 = make_frag(qa0[0], qa0[1], qa0[4], qa0[5], sa0);
    Frag a1 = make_frag(qa1[0], qa1[1], qa1[4], qa1[5], sa1);
    Frag b0 = make_frag(qb0[0], qb0[1], qb0[4], qb0[5], sb0);
    Frag b1 = make_frag(qb1[0], qb1[1], qb1[4], qb1[5], sb1);
    acc00 = wmma3(a0, b0, acc00);
    acc01 = wmma3(a0, b1, acc01);
    acc10 = wmma3(a1, b0, acc10);
    acc11 = wmma3(a1, b1, acc11);
  }

  __shared__ float rsA[4][32];
  __shared__ float rsB[4][32];
  if (mode == 0) {
    // Each row's K-sum is split across lane l and lane l+16; combine with shfl.
    float ra0 = sa0 + __shfl_xor(sa0, 16, 32);
    float ra1 = sa1 + __shfl_xor(sa1, 16, 32);
    float rb0 = sb0 + __shfl_xor(sb0, 16, 32);
    float rb1 = sb1 + __shfl_xor(sb1, 16, 32);
    if (lane < 16) {
      rsA[wid][mrow]      = ra0 * scale;   // scale == 1/HW -> mean
      rsA[wid][16 + mrow] = ra1 * scale;
      rsB[wid][mrow]      = rb0 * scale;
      rsB[wid][16 + mrow] = rb1 * scale;
    }
    __syncthreads();
    if (blockIdx.y == 0 && (wid & 1) == 0 && lane < 16 && aux) {
      aux[b * CC + bm + mrow]      = rsA[wid][mrow];
      aux[b * CC + bm + 16 + mrow] = rsA[wid][16 + mrow];
    }
  }
  const float bscale = (mode == 3) ? aux[b] : scale;

  // C/D layout: VGPR j -> row (j + 8*hi16), col = lane&15.
  const int col   = lane & 15;
  const int rbase = hi16 * 8;
  v8f accs[2][2] = {acc00, acc01, acc10, acc11};
#pragma unroll
  for (int mi = 0; mi < 2; ++mi) {
#pragma unroll
    for (int ni = 0; ni < 2; ++ni) {
      v8f acc = accs[mi][ni];
      const int gn = bn + 16 * ni + col;
#pragma unroll
      for (int j = 0; j < 8; ++j) {
        const int gm = bm + 16 * mi + rbase + j;
        float v = acc[j] * bscale;
        if (mode == 0) {
          float mm = rsA[wid][16 * mi + rbase + j];
          float mn = rsB[wid][16 * ni + col];
          v = v - mm * mn + ((gm == gn) ? EPSV : 0.f);
        } else if (mode == 1) {
          v = 0.5f * (((gm == gn) ? 3.f : 0.f) - v);
        }
        Cb[(long)gm * CC + gn] = v;
      }
    }
  }
}

__global__ __launch_bounds__(128)
void gemm_nt_kernel(const float* __restrict__ A, const float* __restrict__ Bt,
                    float* __restrict__ C, float* __restrict__ aux,
                    int K, int lda, long strideA, long strideB,
                    float scale, int mode)
{
  const int b = blockIdx.z;
  gemm_body(A + (long)b * strideA, Bt + (long)b * strideB,
            C + (long)b * ((long)CC * CC), aux, b, K, lda, scale, mode);
}

// Two independent plain (mode 2) GEMMs in one launch: grid.z in [0,16),
// low 3 bits = batch, bit 3 selects which GEMM. Used for the NS pair
// (y' = y@t, z' = t@z) to halve the dependent-launch chain.
__global__ __launch_bounds__(128)
void gemm_nt_pair_kernel(const float* __restrict__ A0, const float* __restrict__ Bt0,
                         float* __restrict__ C0,
                         const float* __restrict__ A1, const float* __restrict__ Bt1,
                         float* __restrict__ C1)
{
  const int b   = blockIdx.z & 7;
  const int op  = blockIdx.z >> 3;
  const long s  = (long)b * CC * CC;
  const float* A  = (op == 0) ? A0  : A1;
  const float* Bt = (op == 0) ? Bt0 : Bt1;
  float*       C  = (op == 0) ? C0  : C1;
  gemm_body(A + s, Bt + s, C + s, nullptr, b, CC, CC, 1.f, 2);
}

__global__ void frob_kernel(const float* __restrict__ M, float* __restrict__ norm) {
  __shared__ float sm[256];
  const int b = blockIdx.x;
  const float* p = M + (long)b * CC * CC;
  float s = 0.f;
  for (long i = threadIdx.x; i < (long)CC * CC; i += 256) { float x = p[i]; s += x * x; }
  sm[threadIdx.x] = s; __syncthreads();
  for (int off = 128; off > 0; off >>= 1) {
    if (threadIdx.x < off) sm[threadIdx.x] += sm[threadIdx.x + off];
    __syncthreads();
  }
  if (threadIdx.x == 0) norm[b] = sqrtf(sm[0]);
}

__global__ void ns_init_kernel(const float* __restrict__ S, const float* __restrict__ norm,
                               float* __restrict__ Y, float* __restrict__ Z) {
  long i = (long)blockIdx.x * 256 + threadIdx.x;
  int b = (int)(i / ((long)CC * CC));
  int r = (int)((i / CC) % CC);
  int c = (int)(i % CC);
  Y[i] = S[i] / norm[b];
  Z[i] = (r == c) ? 1.f : 0.f;
}

__global__ void final_kernel(const float* __restrict__ meanIn, const float* __restrict__ meanTg,
                             const float* __restrict__ cov, const float* __restrict__ tcov,
                             const float* __restrict__ yfin, const float* __restrict__ norm2,
                             float* __restrict__ out) {
  __shared__ float sm[256];
  const int tid = threadIdx.x;
  float s = 0.f;
  for (int i = tid; i < BB * CC; i += 256) {
    float d = meanIn[i] - meanTg[i];
    s += d * d;
  }
  for (int i = tid; i < BB * CC; i += 256) {
    int b = i >> 9, c = i & (CC - 1);
    long d = (long)b * CC * CC + (long)c * (CC + 1);
    // sqrt_term diag = y_final * sqrt(||M||_F)
    s += tcov[d] + cov[d] - 2.f * sqrtf(norm2[b]) * yfin[d];
  }
  sm[tid] = s; __syncthreads();
  for (int off = 128; off > 0; off >>= 1) {
    if (tid < off) sm[tid] += sm[tid + off];
    __syncthreads();
  }
  if (tid == 0) out[0] = sm[0] * (1.f / (BB * CC));
}

extern "C" void kernel_launch(void* const* d_in, const int* in_sizes, int n_in,
                              void* d_out, int out_size, void* d_ws, size_t ws_size,
                              hipStream_t stream) {
  const float* in = (const float*)d_in[0];
  const float* tg = (const float*)d_in[1];
  float* out = (float*)d_out;
  float* ws  = (float*)d_ws;

  const long BM = (long)BB * CC * CC;      // 2,097,152 floats per matrix set
  float* meanIn = ws;                      // 4096
  float* meanTg = ws + 4096;               // 4096
  float* normA  = ws + 8192;               // 16
  float* normB  = ws + 8208;               // 16
  float* mats   = ws + 8224;
  float* cov  = mats + 0 * BM;
  float* tcov = mats + 1 * BM;
  float* Y0   = mats + 2 * BM;
  float* Y1   = mats + 3 * BM;
  float* Z0   = mats + 4 * BM;
  float* Z1   = mats + 5 * BM;
  float* T    = mats + 6 * BM;
  (void)in_sizes; (void)n_in; (void)out_size; (void)ws_size;

  dim3 gg(CC / 64, CC / 64, BB);           // 8 x 8 x 8
  dim3 gp(CC / 64, CC / 64, 2 * BB);       // paired: 8 x 8 x 16
  dim3 bg(128);
  const long sX = (long)CC * HW_;          // per-batch stride of X
  const long sM = (long)CC * CC;           // per-batch stride of 512x512 matrices
  const int  eBlocks = (int)((BB * sM) / 256); // 8192

  // Gram matrices with fused channel means + (srm - mu mu^T + eps I) epilogue.
  gemm_nt_kernel<<<gg, bg, 0, stream>>>(in, in, cov,  meanIn, HW_, HW_, sX, sX, 1.0f / HW_, 0);
  gemm_nt_kernel<<<gg, bg, 0, stream>>>(tg, tg, tcov, meanTg, HW_, HW_, sX, sX, 1.0f / HW_, 0);

  // sqrtm_ns(target_cov)
  frob_kernel<<<BB, 256, 0, stream>>>(tcov, normA);
  ns_init_kernel<<<eBlocks, 256, 0, stream>>>(tcov, normA, Y0, Z0);
  float *yc = Y0, *ya = Y1, *zc = Z0, *za = Z1;
  for (int it = 0; it < 12; ++it) {
    gemm_nt_kernel<<<gg, bg, 0, stream>>>(zc, yc, T, nullptr, CC, CC, sM, sM, 1.f, 1); // t = 0.5(3I - z y)
    gemm_nt_pair_kernel<<<gp, bg, 0, stream>>>(yc, T, ya,   // y' = y t
                                               T, zc, za);  // z' = t z
    float* tmp;
    tmp = yc; yc = ya; ya = tmp;
    tmp = zc; zc = za; za = tmp;
  }
  // 12 iterations (even) -> yc == Y0 again; tcs = sqrt(normA[b]) * yc (folded below).

  // M = tcs @ cov @ tcs = normA[b] * (y @ cov @ y)   (cov, y symmetric)
  float* P  = Y1; // free (== ya)
  float* Mm = Z1; // free (== za)
  gemm_nt_kernel<<<gg, bg, 0, stream>>>(yc, cov, P, nullptr, CC, CC, sM, sM, 1.f, 2);
  gemm_nt_kernel<<<gg, bg, 0, stream>>>(P,  yc, Mm, normA,   CC, CC, sM, sM, 1.f, 3);

  // sqrtm_ns(M): only its (scaled) diagonal is consumed by the loss.
  frob_kernel<<<BB, 256, 0, stream>>>(Mm, normB);
  ns_init_kernel<<<eBlocks, 256, 0, stream>>>(Mm, normB, Y0, Z0);
  yc = Y0; ya = Y1; zc = Z0; za = Z1;  // first iter overwrites P (Y1) and Mm (Z1): both consumed
  for (int it = 0; it < 12; ++it) {
    gemm_nt_kernel<<<gg, bg, 0, stream>>>(zc, yc, T, nullptr, CC, CC, sM, sM, 1.f, 1);
    gemm_nt_pair_kernel<<<gp, bg, 0, stream>>>(yc, T, ya,
                                               T, zc, za);
    float* tmp;
    tmp = yc; yc = ya; ya = tmp;
    tmp = zc; zc = za; za = tmp;
  }

  final_kernel<<<1, 256, 0, stream>>>(meanIn, meanTg, cov, tcov, yc, normB, out);
}
